// BasicGCNLayer_712964571488
// MI455X (gfx1250) — compile-verified, matrix operand order checked by reading, not run on hardware
//
#include <hip/hip_runtime.h>
#include <hip/hip_bf16.h>

typedef __bf16 bf16;
typedef bf16  v16bf __attribute__((ext_vector_type(16)));
typedef bf16  v8bf  __attribute__((ext_vector_type(8)));
typedef float v8f   __attribute__((ext_vector_type(8)));
typedef int   v4i   __attribute__((ext_vector_type(4)));

#define KB    64     // K-chunk: two WMMA K-steps of 32
#define MB    64     // M rows per block
#define LDSA  72     // bf16 elems per LDS row: 64 used + 8 pad (144B pitch, 16B aligned, conflict-free)

#if __has_builtin(__builtin_amdgcn_global_load_async_to_lds_b128)
#define HAVE_ASYNC_LDS 1
#else
#define HAVE_ASYNC_LDS 0
#endif

// async builtin takes address-space-qualified int4 pointers (per hipcc diagnostic)
#define ASYNC_GPTR(p) ((__attribute__((address_space(1))) v4i*)(p))
#define ASYNC_LPTR(p) ((__attribute__((address_space(3))) v4i*)(p))

__device__ __forceinline__ void wait_async_zero() {
#if HAVE_ASYNC_LDS
#if __has_builtin(__builtin_amdgcn_s_wait_asynccnt)
    __builtin_amdgcn_s_wait_asynccnt(0);
#else
    asm volatile("s_wait_asynccnt 0x0" ::: "memory");
#endif
#endif
}

// ---------------------------------------------------------------------------
// Kernel 1: dinv[i] = rsqrt(1 + sum_j A[i,j])      (one block per row)
// ---------------------------------------------------------------------------
__global__ __launch_bounds__(256) void deg_kernel(const float* __restrict__ A,
                                                  float* __restrict__ dinv, int n) {
    const int row = blockIdx.x;
    const float* a = A + (size_t)row * n;
    float s = 0.0f;
    for (int k = threadIdx.x * 4; k < n; k += blockDim.x * 4) {
        float4 v = *(const float4*)(a + k);
        s += v.x + v.y + v.z + v.w;
    }
    for (int off = 16; off > 0; off >>= 1) s += __shfl_down(s, off, 32);
    __shared__ float wsum[8];
    const int lane = threadIdx.x & 31, wid = threadIdx.x >> 5;
    if (lane == 0) wsum[wid] = s;
    __syncthreads();
    if (threadIdx.x == 0) {
        float d = 1.0f;                        // +I contributes 1 to every row sum
        for (int i = 0; i < 8; ++i) d += wsum[i];
        dinv[row] = d > 0.0f ? rsqrtf(d) : 0.0f;
    }
}

// ---------------------------------------------------------------------------
// Kernel 2: Bt[nn][j] = bf16( dinv[j] * (H[j,:] . W[nn,:] + b[nn]) )   [128][n]
// ---------------------------------------------------------------------------
__global__ __launch_bounds__(256) void linear_kernel(const float* __restrict__ H,
                                                     const float* __restrict__ W,
                                                     const float* __restrict__ b,
                                                     const float* __restrict__ dinv,
                                                     bf16* __restrict__ Bt, int n) {
    const int idx = blockIdx.x * blockDim.x + threadIdx.x;   // < n*128
    const int j  = idx >> 7;
    const int nn = idx & 127;
    const float4* h = (const float4*)(H + (size_t)j * 128);
    const float4* w = (const float4*)(W + (size_t)nn * 128);
    float acc = b[nn];
#pragma unroll 8
    for (int k = 0; k < 32; ++k) {
        float4 hv = h[k], wv = w[k];
        acc += hv.x * wv.x + hv.y * wv.y + hv.z * wv.z + hv.w * wv.w;
    }
    Bt[(size_t)nn * n + j] = (bf16)(dinv[j] * acc);
}

// ---------------------------------------------------------------------------
// Kernel 3: out = relu( Dinv_row ⊙ ( (A+I)_bf16 @ Bt^T ) )
//   256 threads = 8 waves; output tile 64(M) x 128(N); double-buffered LDS;
//   B tile staged via GLOBAL_LOAD_ASYNC_TO_LDS_B128 when available.
// ---------------------------------------------------------------------------
__global__ __launch_bounds__(256) void gcn_main(const float* __restrict__ A,
                                                const bf16* __restrict__ Bt,
                                                const float* __restrict__ dinv,
                                                float* __restrict__ out, int n) {
    __shared__ bf16 sA[2][MB  * LDSA];    // 2 x 64 x 64  (pitch 72) -> 18 KB
    __shared__ bf16 sB[2][128 * LDSA];    // 2 x 128 x 64 (pitch 72) -> 36 KB

    const int tid   = threadIdx.x;
    const int lane  = tid & 31;
    const int wid   = tid >> 5;
    const int mwave = wid & 3;            // 16-row strip of the 64-row tile
    const int nhalf = wid >> 2;           // 64-col half of N=128
    const int m0    = blockIdx.x * MB;
    const int lm    = lane & 15;
    const int lh    = lane >> 4;          // lane half -> K sub-block per ISA layout

    // per-thread staging coordinates (loop-invariant)
    const int ar  = tid >> 5;             // A: base row step (8 rows / iter)
    const int ac  = (tid & 31) * 2;       // A: col pair 0..62
    const int bnr = tid >> 3;             // B: base row step (32 rows / iter)
    const int bch = tid & 7;              // B: 16-byte chunk 0..7

    v8f acc[4] = {v8f{0.f}, v8f{0.f}, v8f{0.f}, v8f{0.f}};

    auto stage = [&](int kc, int buf) {
        // ---- A tile: 64x64 fp32 -> bf16 (+identity), float2 per thread per iter
#pragma unroll
        for (int it = 0; it < 8; ++it) {
            const int r = it * 8 + ar;                     // 0..63
            const float2 v = *(const float2*)(A + (size_t)(m0 + r) * n + kc + ac);
            float a0 = v.x, a1 = v.y;
            const int gr = m0 + r, gc = kc + ac;
            if (gr == gc)     a0 += 1.0f;                  // A + I (exact in bf16)
            if (gr == gc + 1) a1 += 1.0f;
            union { bf16 h[2]; unsigned u; } p;
            p.h[0] = (bf16)a0; p.h[1] = (bf16)a1;
            *(unsigned*)&sA[buf][r * LDSA + ac] = p.u;
        }
        // ---- B tile: 128x64 bf16, 16B chunks
#pragma unroll
        for (int it = 0; it < 4; ++it) {
            const int nr = it * 32 + bnr;                  // 0..127
            const bf16* gsrc = Bt + (size_t)nr * n + kc + bch * 8;
            bf16* ldst = &sB[buf][nr * LDSA + bch * 8];
#if HAVE_ASYNC_LDS
            __builtin_amdgcn_global_load_async_to_lds_b128(
                ASYNC_GPTR(gsrc), ASYNC_LPTR(ldst), 0, 0);
#else
            *(uint4*)ldst = *(const uint4*)gsrc;
#endif
        }
    };

    stage(0, 0);
    const int nch = n / KB;                                // 128 iterations
    for (int i = 0; i < nch; ++i) {
        const int buf = i & 1;
        wait_async_zero();                                 // own async B stores done
        __syncthreads();                                   // tile `buf` visible to all
        if (i + 1 < nch) stage((i + 1) * KB, buf ^ 1);

#pragma unroll
        for (int ks = 0; ks < KB; ks += 32) {
            const int arow = mwave * 16 + lm;
            v8bf alo = *(v8bf*)&sA[buf][arow * LDSA + ks + 8 * lh];
            v8bf ahi = *(v8bf*)&sA[buf][arow * LDSA + ks + 8 * lh + 16];
            v16bf afrag = __builtin_shufflevector(alo, ahi,
                            0,1,2,3,4,5,6,7,8,9,10,11,12,13,14,15);
#pragma unroll
            for (int nt = 0; nt < 4; ++nt) {
                const int brow = nhalf * 64 + nt * 16 + lm;
                v8bf blo = *(v8bf*)&sB[buf][brow * LDSA + ks + 8 * lh];
                v8bf bhi = *(v8bf*)&sB[buf][brow * LDSA + ks + 8 * lh + 16];
                v16bf bfrag = __builtin_shufflevector(blo, bhi,
                                0,1,2,3,4,5,6,7,8,9,10,11,12,13,14,15);
                acc[nt] = __builtin_amdgcn_wmma_f32_16x16x32_bf16(
                              false, afrag, false, bfrag,
                              (short)0, acc[nt], false, false);
            }
        }
        // next loop-top barrier separates compute(buf) from restaging buf
    }

    // ---- epilogue: row scale by dinv, ReLU. C/D layout: M = r + 8*lh, N = lm.
    float rs[8];
#pragma unroll
    for (int r = 0; r < 8; ++r) rs[r] = dinv[m0 + mwave * 16 + 8 * lh + r];
#pragma unroll
    for (int nt = 0; nt < 4; ++nt) {
        const int col = nhalf * 64 + nt * 16 + lm;
#pragma unroll
        for (int r = 0; r < 8; ++r) {
            const int row = m0 + mwave * 16 + 8 * lh + r;
            float v = acc[nt][r] * rs[r];
            out[(size_t)row * 128 + col] = v > 0.0f ? v : 0.0f;
        }
    }
}

// ---------------------------------------------------------------------------
extern "C" void kernel_launch(void* const* d_in, const int* in_sizes, int n_in,
                              void* d_out, int out_size, void* d_ws, size_t ws_size,
                              hipStream_t stream) {
    const float* H = (const float*)d_in[0];   // [N,128]
    const float* A = (const float*)d_in[1];   // [N,N]
    const float* W = (const float*)d_in[2];   // [128,128]
    const float* b = (const float*)d_in[3];   // [128]
    float* out = (float*)d_out;               // [N,128] fp32

    const int N = in_sizes[0] / 128;          // 8192

    float* dinv = (float*)d_ws;
    bf16*  Bt   = (bf16*)((char*)d_ws + (size_t)N * sizeof(float));

    deg_kernel   <<<N,               256, 0, stream>>>(A, dinv, N);
    linear_kernel<<<(N * 128) / 256, 256, 0, stream>>>(H, W, b, dinv, Bt, N);
    gcn_main     <<<N / MB,          256, 0, stream>>>(A, Bt, dinv, out, N);
}
